// MambaBlock_884763263379
// MI455X (gfx1250) — compile-verified, matrix operand order checked by reading, not run on hardware
//
#include <hip/hip_runtime.h>
#include <cstdint>
#include <cstddef>

// ---------------- problem constants ----------------
#define DMODEL   256
#define DSTATE   16
#define DCONV    4
#define DINNER   512
#define DTRANK   16
#define BSZ      2
#define SEQL     8192
#define MROWS    (BSZ * SEQL)       // 16384 token rows
#define XDBL_LD  64                 // x_dbl padded from 48 -> 64 cols
#define LDP      40                 // LDS tile pitch (bf16 elems), 32 data + 8 pad

typedef __attribute__((ext_vector_type(16))) __bf16 v16bf;
typedef __attribute__((ext_vector_type(8)))  float  v8f;
typedef unsigned short bf16_t;

typedef unsigned int u32x4 __attribute__((ext_vector_type(4)));
typedef int          i32x4 __attribute__((ext_vector_type(4)));
typedef int          i32x8 __attribute__((ext_vector_type(8)));

// Tensor Data Mover path: builtin exists on both probed toolchains but with
// different arity (clang-22: 5 args, clang-23: 6 args).
#if defined(__HIP_DEVICE_COMPILE__) && __has_builtin(__builtin_amdgcn_tensor_load_to_lds)
#define USE_TDM 1
#else
#define USE_TDM 0
#endif

// f32 -> bf16 (round to nearest even), bit-level to avoid __bf16 scalar math
__device__ __forceinline__ bf16_t f2bf(float x) {
  unsigned u = __float_as_uint(x);
  unsigned r = 0x7FFFu + ((u >> 16) & 1u);
  return (bf16_t)((u + r) >> 16);
}

struct Frag { unsigned u[8]; };

// Load a 16x32 bf16 fragment from an LDS tile (row pitch LDP elems) using the
// CDNA5 16-bit A/B-matrix lane layout (ISA 7.12.2): lanes 0-15 hold row r,
// K=0..7,16..23; lanes 16-31 hold row r, K=8..15,24..31; 2 elems per VGPR.
__device__ __forceinline__ Frag frag_ld(const bf16_t* base, int ln) {
  int r  = ln & 15;
  int kh = (ln >> 4) << 2;                      // +4 dwords (K+8) for hi half
  const unsigned* p = (const unsigned*)(base + r * LDP);
  Frag f;
#pragma unroll
  for (int j = 0; j < 4; ++j) f.u[j]     = p[j + kh];        // K 0..7   (+kh)
#pragma unroll
  for (int j = 0; j < 4; ++j) f.u[4 + j] = p[8 + j + kh];    // K 16..23 (+kh)
  return f;
}

__device__ __forceinline__ v8f wmma_bf16(Frag a, Frag b, v8f c) {
  v16bf av = __builtin_bit_cast(v16bf, a);
  v16bf bv = __builtin_bit_cast(v16bf, b);
  return __builtin_amdgcn_wmma_f32_16x16x32_bf16(false, av, false, bv,
                                                 (short)0, c, false, false);
}

#if USE_TDM
// LDS aperture: low 32 bits of the generic address are the LDS byte offset.
__device__ __forceinline__ unsigned lds_off(const void* p) {
  return (unsigned)(uintptr_t)p;
}

// Issue one TDM 2D tile load: rows x 32 bf16 tile from a row-major matrix
// (row stride lda elems) into LDS with hardware row padding so the landed
// tile has pitch LDP=40 bf16 (pad_interval=3 -> every 16 DW, pad_amount=3 ->
// 4 DW pad).  D# packing per CDNA5 ISA.
__device__ __forceinline__ void tdm_load_tile(const bf16_t* gptr,
                                              unsigned lds_byte_off,
                                              int rows, int lda) {
  unsigned long long ga = (unsigned long long)(uintptr_t)gptr;
  u32x4 g0;
  g0[0] = 1u;                                       // count=1, no gather
  g0[1] = lds_byte_off;                             // lds_addr
  g0[2] = (unsigned)(ga & 0xFFFFFFFFu);             // global_addr[31:0]
  g0[3] = (unsigned)((ga >> 32) & 0x01FFFFFFu)      // global_addr[56:32]
          | (2u << 30);                             // type = 2 (image)
  i32x8 g1;
  g1[0] = (int)((1u << 16)      // data_size = 1 -> 2 bytes
                | (1u << 20)    // pad_enable
                | (3u << 22)    // pad_interval = 16 DWORDs (one 64B row)
                | (3u << 25));  // pad_amount   = 4 DWORDs (8 bf16)
  g1[1] = (int)(32u << 16);                         // tensor_dim0 = 32 (lo16)
  g1[2] = (int)(((unsigned)rows & 0xFFFFu) << 16);  // tensor_dim1 lo16
  g1[3] = (int)(32u << 16);                         // tile_dim0 = 32
  g1[4] = (int)((unsigned)rows & 0xFFFFu);          // tile_dim1 = rows
  g1[5] = (int)(unsigned)lda;                       // tensor_dim0_stride lo32
  g1[6] = 0;
  g1[7] = 0;
  i32x4 z4 = {0, 0, 0, 0};
#if defined(__clang_major__) && (__clang_major__ >= 23)
  i32x8 z8 = {0, 0, 0, 0, 0, 0, 0, 0};
  __builtin_amdgcn_tensor_load_to_lds(g0, g1, z4, z4, z8, 0);
#else
  __builtin_amdgcn_tensor_load_to_lds(g0, g1, z4, z4, 0);
#endif
}
#endif  // USE_TDM

// ---------------- K0: RMSNorm + bf16 convert ----------------
__global__ __launch_bounds__(256) void rmsnorm_bf16_kernel(
    const float* __restrict__ x, const float* __restrict__ w,
    bf16_t* __restrict__ out) {
  int row = blockIdx.x;            // token row
  int tid = threadIdx.x;           // 0..255 (== DMODEL)
  float v = x[(size_t)row * DMODEL + tid];
  float s = v * v;
#pragma unroll
  for (int off = 16; off > 0; off >>= 1) s += __shfl_xor(s, off, 32);
  __shared__ float ssum[8];
  if ((tid & 31) == 0) ssum[tid >> 5] = s;
  __syncthreads();
  float tot = 0.f;
#pragma unroll
  for (int i = 0; i < 8; ++i) tot += ssum[i];
  float r = rsqrtf(tot * (1.0f / DMODEL) + 1e-5f);
  out[(size_t)row * DMODEL + tid] = f2bf(v * r * w[tid]);
}

// ---------------- weight convert (with optional zero padding) -------------
__global__ __launch_bounds__(256) void cvt_bf16_kernel(
    const float* __restrict__ src, bf16_t* __restrict__ dst,
    int n_src, int n_total) {
  int i = blockIdx.x * 256 + threadIdx.x;
  if (i < n_total) dst[i] = (i < n_src) ? f2bf(src[i]) : (bf16_t)0;
}

// ---------------- generic bf16 WMMA GEMM: C = A * W^T ----------------
// A: MxK row-major (bf16), W: NxK row-major (bf16).  MODE epilogues:
//   0: split store -> out0 = xs (cols 0..511), out1 = z (cols 512..1023)
//   1: plain f32 store, leading dim ldo
//   2: out0[gm*ldo + coloff + gn] = acc + resid[gm*DMODEL + gn]
template <int BM, int BN, int WM, int WN, int MODE>
__global__ __launch_bounds__(256) void gemm_bf16_kernel(
    const bf16_t* __restrict__ A, int lda,
    const bf16_t* __restrict__ W, int ldw,
    int K,
    float* __restrict__ out0, float* __restrict__ out1,
    const float* __restrict__ resid, int ldo, int coloff) {
  constexpr int MT = WM / 16;
  constexpr int NT = WN / 16;
  static_assert((BM / WM) * (BN / WN) * 32 == 256, "8 waves of 32");

  __shared__ __align__(16) bf16_t As[2][BM * LDP];
  __shared__ __align__(16) bf16_t Ws[2][BN * LDP];

  const int tid  = threadIdx.x;
  const int wave = tid >> 5;
  const int ln   = tid & 31;
  const int m0   = blockIdx.y * BM;
  const int n0   = blockIdx.x * BN;
  const int wm   = (wave % (BM / WM)) * WM;
  const int wn   = (wave / (BM / WM)) * WN;
  const int nk   = K / 32;

  v8f acc[MT][NT] = {};

#if USE_TDM
  // TDM-fed, double-buffered pipeline: DMA of tile k+1 overlaps WMMA on k.
  const bf16_t* Abase = A + (size_t)m0 * lda;
  const bf16_t* Wbase = W + (size_t)n0 * ldw;
  if (wave == 0) {
    tdm_load_tile(Abase, lds_off(&As[0][0]), BM, lda);
    tdm_load_tile(Wbase, lds_off(&Ws[0][0]), BN, ldw);
  }
  int buf = 0;
  for (int ks = 0; ks < nk; ++ks) {
    if (wave == 0) __builtin_amdgcn_s_wait_tensorcnt(0);
    __syncthreads();   // tiles in buf ready; all waves past previous compute
    if (wave == 0 && ks + 1 < nk) {
      tdm_load_tile(Abase + (ks + 1) * 32, lds_off(&As[buf ^ 1][0]), BM, lda);
      tdm_load_tile(Wbase + (ks + 1) * 32, lds_off(&Ws[buf ^ 1][0]), BN, ldw);
    }
    const bf16_t* at = As[buf];
    const bf16_t* wt = Ws[buf];
    Frag af[MT], bf[NT];
#pragma unroll
    for (int i = 0; i < MT; ++i) af[i] = frag_ld(at + (wm + i * 16) * LDP, ln);
#pragma unroll
    for (int j = 0; j < NT; ++j) bf[j] = frag_ld(wt + (wn + j * 16) * LDP, ln);
#pragma unroll
    for (int i = 0; i < MT; ++i)
#pragma unroll
      for (int j = 0; j < NT; ++j)
        acc[i][j] = wmma_bf16(af[i], bf[j], acc[i][j]);
    buf ^= 1;
  }
#else
  // fallback: manual VMEM -> LDS staging
  for (int k0 = 0; k0 < K; k0 += 32) {
    __syncthreads();
    for (int idx = tid; idx < BM * 4; idx += 256) {
      int r = idx >> 2, s = idx & 3;
      uint4 d = *(const uint4*)(A + (size_t)(m0 + r) * lda + k0 + s * 8);
      *(uint4*)(&As[0][0] + r * LDP + s * 8) = d;
    }
    for (int idx = tid; idx < BN * 4; idx += 256) {
      int r = idx >> 2, s = idx & 3;
      uint4 d = *(const uint4*)(W + (size_t)(n0 + r) * ldw + k0 + s * 8);
      *(uint4*)(&Ws[0][0] + r * LDP + s * 8) = d;
    }
    __syncthreads();
    Frag af[MT], bf[NT];
#pragma unroll
    for (int i = 0; i < MT; ++i)
      af[i] = frag_ld(&As[0][0] + (wm + i * 16) * LDP, ln);
#pragma unroll
    for (int j = 0; j < NT; ++j)
      bf[j] = frag_ld(&Ws[0][0] + (wn + j * 16) * LDP, ln);
#pragma unroll
    for (int i = 0; i < MT; ++i)
#pragma unroll
      for (int j = 0; j < NT; ++j)
        acc[i][j] = wmma_bf16(af[i], bf[j], acc[i][j]);
  }
#endif

  // epilogue: C/D layout (ISA 7.12.2): lane -> col n = ln&15,
  // rows (ln>>4)*8 + r across the 8 accumulator dwords
  const int cn = ln & 15;
  const int mh = (ln >> 4) * 8;
#pragma unroll
  for (int i = 0; i < MT; ++i)
#pragma unroll
    for (int j = 0; j < NT; ++j)
#pragma unroll
      for (int r = 0; r < 8; ++r) {
        int gm = m0 + wm + i * 16 + mh + r;
        int gn = n0 + wn + j * 16 + cn;
        float v = acc[i][j][r];
        if (MODE == 0) {
          if (gn < DINNER) out0[(size_t)gm * DINNER + gn] = v;
          else             out1[(size_t)gm * DINNER + gn - DINNER] = v;
        } else if (MODE == 1) {
          out0[(size_t)gm * ldo + gn] = v;
        } else {
          out0[(size_t)gm * ldo + coloff + gn] =
              v + resid[(size_t)gm * DMODEL + gn];
        }
      }
}

// ---------------- K2: depthwise causal conv (dir-aware) + SiLU ------------
__global__ __launch_bounds__(256) void conv_silu_kernel(
    const float* __restrict__ xs, const float* __restrict__ w,
    const float* __restrict__ b, float* __restrict__ xc,
    bf16_t* __restrict__ xcb, int rev) {
  size_t idx = (size_t)blockIdx.x * 256 + threadIdx.x;   // b,t,d flattened
  int d  = (int)(idx % DINNER);
  int t  = (int)((idx / DINNER) % SEQL);
  int bb = (int)(idx / ((size_t)DINNER * SEQL));
  const float* wd = w + d * DCONV;
  float a = 0.f;
  if (!rev) {
#pragma unroll
    for (int k = 0; k < DCONV; ++k) {
      int tt = t - (DCONV - 1) + k;
      if (tt >= 0) a += wd[k] * xs[((size_t)bb * SEQL + tt) * DINNER + d];
    }
  } else {  // time-flipped causal conv expressed in original coordinates
#pragma unroll
    for (int k = 0; k < DCONV; ++k) {
      int tt = t + (DCONV - 1) - k;
      if (tt < SEQL) a += wd[k] * xs[((size_t)bb * SEQL + tt) * DINNER + d];
    }
  }
  a += b[d];
  float s = a / (1.f + __expf(-a));   // SiLU
  xc[idx]  = s;
  xcb[idx] = f2bf(s);
}

// ---------------- K4: dt = softplus(x_dbl[:, :16] @ dt_w^T + dt_b) --------
__global__ __launch_bounds__(256) void dt_softplus_kernel(
    const float* __restrict__ xdbl, const float* __restrict__ dtw,
    const float* __restrict__ dtb, float* __restrict__ dt) {
  size_t idx = (size_t)blockIdx.x * 256 + threadIdx.x;
  int d = (int)(idx % DINNER);
  size_t m = idx / DINNER;
  const float* xr = xdbl + m * XDBL_LD;
  const float* wr = dtw + d * DTRANK;
  float a = 0.f;
#pragma unroll
  for (int r = 0; r < DTRANK; ++r) a += xr[r] * wr[r];
  a += dtb[d];
  float sp = (a > 20.f) ? a : log1pf(__expf(a));
  dt[idx] = sp;
}

// ---------------- K5: selective scan ----------------
// wave32 layout: lane = ds*16 + n ; each wave owns (batch, 2 channels) and
// all 16 states; y_t = sum_n h*C via 16-lane shfl_xor reduction.
__global__ __launch_bounds__(256) void scan_kernel(
    const float* __restrict__ dt, const float* __restrict__ xc,
    const float* __restrict__ xdbl, const float* __restrict__ A_log,
    float* __restrict__ ys, int rev) {
  int w  = (blockIdx.x * 256 + threadIdx.x) >> 5;   // 0..511
  int ln = threadIdx.x & 31;
  int b  = w >> 8;                 // batch
  int dp = w & 255;                // channel pair
  int n  = ln & 15;
  int d  = dp * 2 + (ln >> 4);
  const float A = -__expf(A_log[d * DSTATE + n]);
  float h = 0.f;
  const int t0 = rev ? (SEQL - 1) : 0;
  const int st = rev ? -1 : 1;
  for (int i = 0; i < SEQL; ++i) {
    int t = t0 + st * i;
    size_t off = (size_t)b * SEQL + t;
    if (i + 8 < SEQL) {  // keep streams warm in L2/L0
      size_t po = (size_t)b * SEQL + (t + st * 8);
      __builtin_prefetch(dt + po * DINNER + d, 0, 0);
      __builtin_prefetch(xdbl + po * XDBL_LD + DTRANK + n, 0, 0);
    }
    float dtv = dt[off * DINNER + d];
    float xv  = xc[off * DINNER + d];
    float Bv  = xdbl[off * XDBL_LD + DTRANK + n];
    float Cv  = xdbl[off * XDBL_LD + DTRANK + DSTATE + n];
    float dA  = __expf(dtv * A);
    h = h * dA + dtv * xv * Bv;
    float p = h * Cv;
    p += __shfl_xor(p, 1, 32);
    p += __shfl_xor(p, 2, 32);
    p += __shfl_xor(p, 4, 32);
    p += __shfl_xor(p, 8, 32);
    if (n == 0) ys[off * DINNER + d] = p;
  }
}

// ---------------- K6a: y = (ys + D*xc) * silu(z), -> bf16 ----------------
__global__ __launch_bounds__(256) void fuse_y_kernel(
    const float* __restrict__ ys, const float* __restrict__ xc,
    const float* __restrict__ Dp, const float* __restrict__ z,
    bf16_t* __restrict__ yb) {
  size_t idx = (size_t)blockIdx.x * 256 + threadIdx.x;
  int d = (int)(idx % DINNER);
  float y  = ys[idx] + Dp[d] * xc[idx];
  float zv = z[idx];
  y *= zv / (1.f + __expf(-zv));
  yb[idx] = f2bf(y);
}

// ---------------- host-side orchestration ----------------
extern "C" void kernel_launch(void* const* d_in, const int* in_sizes, int n_in,
                              void* d_out, int out_size, void* d_ws,
                              size_t ws_size, hipStream_t stream) {
  const float* x = (const float*)d_in[0];

  // workspace carving (reused across the two sequential directions)
  char* p = (char*)d_ws;
  auto alloc = [&](size_t bytes) {
    void* r = (void*)p;
    p += (bytes + 255) & ~(size_t)255;
    return r;
  };
  bf16_t* hnorm_bf = (bf16_t*)alloc((size_t)MROWS * DMODEL * 2);
  bf16_t* w_in_bf  = (bf16_t*)alloc((size_t)(2 * DINNER) * DMODEL * 2);
  bf16_t* w_xp_bf  = (bf16_t*)alloc((size_t)XDBL_LD * DINNER * 2);   // padded
  bf16_t* w_out_bf = (bf16_t*)alloc((size_t)DMODEL * DINNER * 2);
  float*  xs_f     = (float*)alloc((size_t)MROWS * DINNER * 4);
  float*  z_f      = (float*)alloc((size_t)MROWS * DINNER * 4);
  float*  xc_f     = (float*)alloc((size_t)MROWS * DINNER * 4);
  bf16_t* xc_bf    = (bf16_t*)alloc((size_t)MROWS * DINNER * 2);
  float*  xdbl_f   = (float*)alloc((size_t)MROWS * XDBL_LD * 4);
  float*  dt_f     = (float*)alloc((size_t)MROWS * DINNER * 4);
  float*  ys_f     = (float*)alloc((size_t)MROWS * DINNER * 4);
  bf16_t* y_bf     = (bf16_t*)alloc((size_t)MROWS * DINNER * 2);

  const int EW_BLOCKS = (MROWS * DINNER) / 256;   // 32768

  for (int dir = 0; dir < 2; ++dir) {
    int o = 1 + dir * 10;
    const float* W_in   = (const float*)d_in[o + 0];
    const float* conv_w = (const float*)d_in[o + 1];
    const float* conv_b = (const float*)d_in[o + 2];
    const float* W_xp   = (const float*)d_in[o + 3];
    const float* dt_w   = (const float*)d_in[o + 4];
    const float* dt_b   = (const float*)d_in[o + 5];
    const float* A_log  = (const float*)d_in[o + 6];
    const float* Dp     = (const float*)d_in[o + 7];
    const float* W_out  = (const float*)d_in[o + 8];
    const float* norm_w = (const float*)d_in[o + 9];

    // weight conversions (x_proj zero-padded 48 -> 64 rows)
    cvt_bf16_kernel<<<(2 * DINNER * DMODEL + 255) / 256, 256, 0, stream>>>(
        W_in, w_in_bf, 2 * DINNER * DMODEL, 2 * DINNER * DMODEL);
    cvt_bf16_kernel<<<(XDBL_LD * DINNER + 255) / 256, 256, 0, stream>>>(
        W_xp, w_xp_bf, (DTRANK + 2 * DSTATE) * DINNER, XDBL_LD * DINNER);
    cvt_bf16_kernel<<<(DMODEL * DINNER + 255) / 256, 256, 0, stream>>>(
        W_out, w_out_bf, DMODEL * DINNER, DMODEL * DINNER);

    // K0: RMSNorm -> bf16 activations
    rmsnorm_bf16_kernel<<<MROWS, 256, 0, stream>>>(x, norm_w, hnorm_bf);

    // K1: xz = h @ in_proj^T  (16384 x 256 x 1024), split xs / z
    gemm_bf16_kernel<128, 128, 64, 32, 0>
        <<<dim3((2 * DINNER) / 128, MROWS / 128), 256, 0, stream>>>(
            hnorm_bf, DMODEL, w_in_bf, DMODEL, DMODEL,
            xs_f, z_f, nullptr, 0, 0);

    // K2: depthwise causal conv (direction-aware) + SiLU
    conv_silu_kernel<<<EW_BLOCKS, 256, 0, stream>>>(xs_f, conv_w, conv_b,
                                                    xc_f, xc_bf, dir);

    // K3: x_dbl = conv(x) @ x_proj^T  (16384 x 512 x 64-padded)
    gemm_bf16_kernel<128, 64, 32, 32, 1>
        <<<dim3(1, MROWS / 128), 256, 0, stream>>>(
            xc_bf, DINNER, w_xp_bf, DINNER, DINNER,
            xdbl_f, nullptr, nullptr, XDBL_LD, 0);

    // K4: dt = softplus(dt_lin + dt_b)
    dt_softplus_kernel<<<EW_BLOCKS, 256, 0, stream>>>(xdbl_f, dt_w, dt_b, dt_f);

    // K5: selective scan (512 waves; reversed walk for backward dir)
    scan_kernel<<<64, 256, 0, stream>>>(dt_f, xc_f, xdbl_f, A_log, ys_f, dir);

    // K6a: y = (ys + D*xc) * silu(z)
    fuse_y_kernel<<<EW_BLOCKS, 256, 0, stream>>>(ys_f, xc_f, Dp, z_f, y_bf);

    // K6b: out = y @ out_proj^T + residual, into d_out cols [dir*256, +256)
    gemm_bf16_kernel<128, 128, 64, 32, 2>
        <<<dim3(DMODEL / 128, MROWS / 128), 256, 0, stream>>>(
            y_bf, DINNER, w_out_bf, DINNER, DINNER,
            (float*)d_out, nullptr, x, 2 * DMODEL, dir * DMODEL);
  }
}
